// DeepCell_37168646980394
// MI455X (gfx1250) — matrix operand order repaired and checked by reading, not compiled, not run on hardware
//
#include <hip/hip_runtime.h>
#include <hip/hip_bf16.h>

typedef __attribute__((ext_vector_type(16))) _Float16 v16h;
typedef __attribute__((ext_vector_type(8)))  _Float16 v8h;
typedef __attribute__((ext_vector_type(8)))  float    v8f;

__constant__ int c_FOFF[13] = {0,6,16,18,19,20,21,22,24,25,26,28,30}; // cumulative FEATURE_SIZES
__constant__ int c_EMB[12]  = {9,12,4,1,1,1,1,4,1,1,4,4};

// ---------------- workspace layout (float offsets) ----------------
#define WS_PROJ  0      // [30][16] fused proj tables (f32)
#define WS_FIRST 480    // [32] first-order scalar table (f32, padded)
#define WS_LINB  512    // [12][16] lin_b (f32)
#define WS_B1    704    // [32]
#define WS_B2    736    // [64]
#define WS_B3    800    // [32]
#define WS_W4    832    // [32] final layer weights (f32)
#define WS_SCAL  864    // [0]=mlp_b4, [1]=global bias
#define WS_HALF  896    // start of f16 region (in float units)
#define NH_W1    6144   // 192x32 f16, B-fragment packed
#define NH_W2    2048   // 32x64
#define NH_W3    2048   // 64x32
#define NH_TOT   (NH_W1+NH_W2+NH_W3)

struct Params {
  const int*   X; const float* w;
  const float* ft[12]; const float* st[12];
  const float* lw[12]; const float* lb[12];
  const float* W[4];   const float* b[4];
  const float* bias;
};

// ---------------- prep: fuse tables + pack WMMA B fragments ----------------
// B-fragment packing for a K x N (k-major) W^T tile set:
//   fragIdx = kt*(N/16)+nt ; element (lane l, half h):
//   k = kt*32 + (l>=16 ? 16 : 0) + h ; n = nt*16 + (l&15)
__global__ void deepfm_prep(Params p, float* ws) {
  _Float16* wh = (_Float16*)(ws + WS_HALF);
  const int tid = threadIdx.x, nthr = blockDim.x;
  for (int t = tid; t < 480; t += nthr) {          // proj[row][e] = sum_j st[row][j]*lw[e][j]
    int row = t >> 4, e = t & 15;
    int i = 0; while (!(row < c_FOFF[i+1])) ++i;
    int v = row - c_FOFF[i], ei = c_EMB[i];
    float acc = 0.f;
    for (int j = 0; j < ei; ++j) acc += p.st[i][v*ei + j] * p.lw[i][e*ei + j];
    ws[WS_PROJ + t] = acc;
  }
  for (int t = tid; t < 32; t += nthr) {
    float v = 0.f;
    if (t < 30) { int i = 0; while (!(t < c_FOFF[i+1])) ++i; v = p.ft[i][t - c_FOFF[i]]; }
    ws[WS_FIRST + t] = v;
  }
  for (int t = tid; t < 192; t += nthr) ws[WS_LINB + t] = p.lb[t >> 4][t & 15];
  for (int t = tid; t < 32;  t += nthr) ws[WS_B1 + t] = p.b[0][t];
  for (int t = tid; t < 64;  t += nthr) ws[WS_B2 + t] = p.b[1][t];
  for (int t = tid; t < 32;  t += nthr) ws[WS_B3 + t] = p.b[2][t];
  for (int t = tid; t < 32;  t += nthr) ws[WS_W4 + t] = p.W[3][t];
  if (tid == 0) { ws[WS_SCAL] = p.b[3][0]; ws[WS_SCAL + 1] = p.bias[0]; }
  for (int t = tid; t < NH_W1; t += nthr) {        // W1: [32,192] -> K=192,N=32
    int h = t & 15, l = (t >> 4) & 31, frag = t >> 9, kt = frag >> 1, nt = frag & 1;
    int k = kt*32 + ((l >> 4) << 4) + h, n = (nt << 4) + (l & 15);
    wh[t] = (_Float16)p.W[0][n*192 + k];
  }
  for (int t = tid; t < NH_W2; t += nthr) {        // W2: [64,32] -> K=32,N=64
    int h = t & 15, l = (t >> 4) & 31, nt = t >> 9;
    int k = ((l >> 4) << 4) + h, n = nt*16 + (l & 15);
    wh[NH_W1 + t] = (_Float16)p.W[1][n*32 + k];
  }
  for (int t = tid; t < NH_W3; t += nthr) {        // W3: [32,64] -> K=64,N=32
    int h = t & 15, l = (t >> 4) & 31, frag = t >> 9, kt = frag >> 1, nt = frag & 1;
    int k = kt*32 + ((l >> 4) << 4) + h, n = nt*16 + (l & 15);
    wh[NH_W1 + NH_W2 + t] = (_Float16)p.W[2][n*64 + k];
  }
}

// ---------------- fragment loaders (contiguous k order: k = 16*(lane/16) + s) ----------------
__device__ __forceinline__ v16h load_a(const _Float16* abuf, int stride, int k0, int lane) {
  // 16-bit A 16x32: lanes 0-15 hold K=[0..15]; lanes 16-31 K=[16..31]
  const _Float16* base = abuf + (lane & 15) * stride + k0 + ((lane >> 4) << 4);
  v8h lo = *(const v8h*)(base);
  v8h hi = *(const v8h*)(base + 8);
  return __builtin_shufflevector(lo, hi, 0,1,2,3,4,5,6,7,8,9,10,11,12,13,14,15);
}
__device__ __forceinline__ v16h load_b(const _Float16* wbuf, int frag, int lane) {
  const _Float16* base = wbuf + ((frag << 5) + lane) * 16;  // packed: 16 contiguous halves/lane
  v8h lo = *(const v8h*)(base);
  v8h hi = *(const v8h*)(base + 8);
  return __builtin_shufflevector(lo, hi, 0,1,2,3,4,5,6,7,8,9,10,11,12,13,14,15);
}

// ---------------- main kernel: 4 waves/block, 16 rows/wave, chunk loop ----------------
__launch_bounds__(128)
__global__ void deepfm_main(const int* __restrict__ X, const float* __restrict__ Wgt,
                            const float* __restrict__ ws, float* __restrict__ out,
                            int n, int chunks) {
  __shared__ float s_proj[480];
  __shared__ float s_first[32];
  __shared__ float s_linb[192];
  __shared__ float s_b1[32], s_b2[64], s_b3[32], s_w4[32], s_scal[2];
  __shared__ __align__(32) _Float16 s_wh[NH_TOT];
  __shared__ __align__(32) _Float16 s_abuf[4][16*192];

  const int tid = threadIdx.x;
  for (int t = tid; t < 480; t += 128) s_proj[t] = ws[WS_PROJ + t];
  for (int t = tid; t < 32;  t += 128) s_first[t] = ws[WS_FIRST + t];
  for (int t = tid; t < 192; t += 128) s_linb[t] = ws[WS_LINB + t];
  for (int t = tid; t < 32;  t += 128) s_b1[t] = ws[WS_B1 + t];
  for (int t = tid; t < 64;  t += 128) s_b2[t] = ws[WS_B2 + t];
  for (int t = tid; t < 32;  t += 128) s_b3[t] = ws[WS_B3 + t];
  for (int t = tid; t < 32;  t += 128) s_w4[t] = ws[WS_W4 + t];
  if (tid < 2) s_scal[tid] = ws[WS_SCAL + tid];
  { const uint32_t* src = (const uint32_t*)(ws + WS_HALF);
    uint32_t* dst = (uint32_t*)s_wh;
    for (int t = tid; t < NH_TOT/2; t += 128) dst[t] = src[t]; }
  __syncthreads();

  const int lane = tid & 31, wave = tid >> 5;
  const int m = lane & 15, half = lane >> 4;
  _Float16* abuf = s_abuf[wave];

  for (int c = blockIdx.x; c < chunks; c += gridDim.x) {
    // prefetch next chunk's gather data (global_prefetch_b8)
    const int cn = c + gridDim.x;
    if (cn < chunks) {
      const char* px = (const char*)(X   + (size_t)cn * 64 * 12);
      const char* pw = (const char*)(Wgt + (size_t)cn * 64 * 12);
      __builtin_prefetch(px + tid * 24, 0, 0);   // 128 thr * 24B = 3072B (int32 block)
      __builtin_prefetch(pw + tid * 24, 0, 0);   // 3072B (f32 block)
    }

    const int r0 = c * 64 + wave * 16;
    const int row = r0 + m;
    const int rowL = row < n ? row : (n - 1);

    // ---- phase 1: gather + fused projection, build A tile [16 x 192] f16 ----
    // lane's 6 fields are contiguous: force b64 vector loads (8B-aligned: 48B row, 24B half)
    const int gbase = rowL * 12 + half * 6;
    int2   xv[3];
    float2 wv2[3];
#pragma unroll
    for (int q = 0; q < 3; ++q) {
      xv[q]  = *(const int2*)  (X   + gbase + q * 2);
      wv2[q] = *(const float2*)(Wgt + gbase + q * 2);
    }
    int   idxs[6] = {xv[0].x, xv[0].y, xv[1].x, xv[1].y, xv[2].x, xv[2].y};
    float wts[6]  = {wv2[0].x, wv2[0].y, wv2[1].x, wv2[1].y, wv2[2].x, wv2[2].y};

    float es[16], ss[16];
#pragma unroll
    for (int e = 0; e < 16; ++e) { es[e] = 0.f; ss[e] = 0.f; }
    float fo = 0.f;
#pragma unroll
    for (int ii = 0; ii < 6; ++ii) {
      const int i = half * 6 + ii;
      const int idx  = idxs[ii];
      const float wv = wts[ii];
      const int trow = c_FOFF[i] + idx;
      fo += wv * s_first[trow];
      const int pb = trow << 4, lb = i << 4;
#pragma unroll
      for (int e = 0; e < 16; ++e) {
        float s = fmaf(wv, s_proj[pb + e], s_linb[lb + e]);
        es[e] += s;
        ss[e] = fmaf(s, s, ss[e]);
        abuf[m * 192 + lb + e] = (_Float16)s;
      }
    }
    const float fo_full = fo + __shfl_xor(fo, 16);
    float fm2 = 0.f;
#pragma unroll
    for (int e = 0; e < 16; ++e) {
      float esF = es[e] + __shfl_xor(es[e], 16);
      float ssF = ss[e] + __shfl_xor(ss[e], 16);
      fm2 += 0.5f * (esF * esF - ssF);
    }
    // abuf is wave-private; per-wave LDS ops are in-order -> no barrier needed.

    // ---- layer 1: [16x192] @ [192x32] -> 12 WMMA (C starts as inline 0, bias folded at ReLU) ----
    v8f c1[2] = {};
#pragma unroll
    for (int kt = 0; kt < 6; ++kt) {
      v16h a = load_a(abuf, 192, kt * 32, lane);
#pragma unroll
      for (int nt = 0; nt < 2; ++nt) {
        v16h b = load_b(s_wh, kt * 2 + nt, lane);
        c1[nt] = __builtin_amdgcn_wmma_f32_16x16x32_f16(false, a, false, b, (short)0, c1[nt], false, false);
      }
    }
#pragma unroll
    for (int nt = 0; nt < 2; ++nt) {
      const float bv = s_b1[m + nt * 16];
#pragma unroll
      for (int r = 0; r < 8; ++r)
        abuf[(r + half * 8) * 32 + nt * 16 + m] = (_Float16)fmaxf(c1[nt][r] + bv, 0.f);
    }

    // ---- layer 2: [16x32] @ [32x64] -> 4 WMMA ----
    v8f c2[4] = {};
    {
      v16h a = load_a(abuf, 32, 0, lane);
#pragma unroll
      for (int nt = 0; nt < 4; ++nt) {
        v16h b = load_b(s_wh + NH_W1, nt, lane);
        c2[nt] = __builtin_amdgcn_wmma_f32_16x16x32_f16(false, a, false, b, (short)0, c2[nt], false, false);
      }
    }
#pragma unroll
    for (int nt = 0; nt < 4; ++nt) {
      const float bv = s_b2[m + nt * 16];
#pragma unroll
      for (int r = 0; r < 8; ++r)
        abuf[(r + half * 8) * 64 + nt * 16 + m] = (_Float16)fmaxf(c2[nt][r] + bv, 0.f);
    }

    // ---- layer 3: [16x64] @ [64x32] -> 4 WMMA ----
    v8f c3[2] = {};
#pragma unroll
    for (int kt = 0; kt < 2; ++kt) {
      v16h a = load_a(abuf, 64, kt * 32, lane);
#pragma unroll
      for (int nt = 0; nt < 2; ++nt) {
        v16h b = load_b(s_wh + NH_W1 + NH_W2, kt * 2 + nt, lane);
        c3[nt] = __builtin_amdgcn_wmma_f32_16x16x32_f16(false, a, false, b, (short)0, c3[nt], false, false);
      }
    }
    float* fbuf = (float*)abuf;  // reuse wave-private buffer as 16x32 f32
#pragma unroll
    for (int nt = 0; nt < 2; ++nt) {
      const float bv = s_b3[m + nt * 16];
#pragma unroll
      for (int r = 0; r < 8; ++r)
        fbuf[(r + half * 8) * 32 + nt * 16 + m] = fmaxf(c3[nt][r] + bv, 0.f);
    }

    // ---- layer 4 (dot with w4) + combine; lanes 0-15 own one row each ----
    if (lane < 16 && row < n) {
      float dot = 0.f;
#pragma unroll
      for (int nn = 0; nn < 32; ++nn) dot = fmaf(fbuf[lane * 32 + nn], s_w4[nn], dot);
      float x0 = fmaxf(dot + s_scal[0], 0.f);
      out[row] = fo_full + fm2 + x0 + s_scal[1];
    }
  }
}

extern "C" void kernel_launch(void* const* d_in, const int* in_sizes, int n_in,
                              void* d_out, int out_size, void* d_ws, size_t ws_size,
                              hipStream_t stream) {
  Params p;
  p.X = (const int*)d_in[0];
  p.w = (const float*)d_in[1];
  for (int i = 0; i < 12; ++i) {
    p.ft[i] = (const float*)d_in[2 + i];
    p.st[i] = (const float*)d_in[14 + i];
    p.lw[i] = (const float*)d_in[26 + i];
    p.lb[i] = (const float*)d_in[38 + i];
  }
  for (int j = 0; j < 4; ++j) {
    p.W[j] = (const float*)d_in[50 + j];
    p.b[j] = (const float*)d_in[54 + j];
  }
  p.bias = (const float*)d_in[58];

  const int n = in_sizes[0] / 12;
  const int chunks = (n + 63) / 64;
  int grid = chunks < 2048 ? chunks : 2048;
  float* ws = (float*)d_ws;

  deepfm_prep<<<1, 256, 0, stream>>>(p, ws);
  deepfm_main<<<grid, 128, 0, stream>>>(p.X, p.w, (const float*)ws,
                                        (float*)d_out, n, chunks);
}
